// ConvSDF_62191126446547
// MI455X (gfx1250) — matrix operand order repaired and checked by reading, not compiled, not run on hardware
//
#include <hip/hip_runtime.h>
#include <hip/hip_bf16.h>

typedef __attribute__((ext_vector_type(16))) _Float16 v16h;
typedef __attribute__((ext_vector_type(8)))  float    v8f;

#define GRID_N   64
#define GRID3    (GRID_N * GRID_N * GRID_N)
#define KTAP     27
#define NOBJ     4
#define NOUT     32
#define DILF     0.05f
#define MAXDF    1.0f
#define WAVES_PB 8

__global__ __launch_bounds__(WAVES_PB * 32)
void ConvSDF_62191126446547_kernel(const float* __restrict__ locs,
                                   const int*   __restrict__ idxs,
                                   const float* __restrict__ poses,
                                   const float* __restrict__ scales,
                                   const float* __restrict__ sdf,
                                   const float* __restrict__ cell_sizes,
                                   const float* __restrict__ weight,
                                   const float* __restrict__ bias,
                                   float*       __restrict__ out,
                                   int npts, int ntiles)
{
    __shared__ _Float16 dstage[WAVES_PB * 32 * 16];   // 8 KB: per-lane A slots

    const int lane = threadIdx.x & 31;
    const int wave = threadIdx.x >> 5;
    const int tile = blockIdx.x * WAVES_PB + wave;
    if (tile >= ntiles) return;                 // uniform per wave -> EXEC stays full

    const int bb   = blockIdx.y;                // batch (uniform per block)
    const int g    = lane >> 4;                 // lane half: K-group select
    const int col  = lane & 15;                 // A row (point) / B,D column
    const int base = tile * 16;

    // ---- per-object params (block-uniform -> scalar loads) ----
    float tx[NOBJ], ty[NOBJ], tz[NOBJ];
    float ux[NOBJ], uy[NOBJ], uz[NOBJ], qw[NOBJ];
    float scl[NOBJ], invs[NOBJ], invc[NOBJ], hib[NOBJ];
    int   gb[NOBJ];
#pragma unroll
    for (int m = 0; m < NOBJ; ++m) {
        const float* P = poses + ((long)bb * NOBJ + m) * 7;
        tx[m] = P[0]; ty[m] = P[1]; tz[m] = P[2];
        ux[m] = -P[3]; uy[m] = -P[4]; uz[m] = -P[5]; qw[m] = P[6];
        float s = scales[bb * NOBJ + m];
        scl[m] = s; invs[m] = 1.0f / s;
        int id = idxs[bb * NOBJ + m];
        float cs = cell_sizes[id];
        invc[m] = 1.0f / cs;
        hib[m]  = (float)GRID_N * cs;
        gb[m]   = id * GRID3;
    }

    int pt = base + col;
    if (pt >= npts) pt = npts - 1;
    const float* L = locs + ((long)bb * npts + pt) * 3;
    const float lx = L[0], ly = L[1], lz = L[2];

    const int myOff = (wave * 32 + lane) * 16;

    // ---- 16 distance taps per lane, matching 16-bit A (16x32) fragment layout:
    //      k = i + (i>=8 ? 8 : 0) + 8*g  (rows = lanes 0..15, K-halves by lane>=16)
#pragma unroll 1
    for (int i = 0; i < 16; ++i) {
        const int k = i + ((i >= 8) ? 8 : 0) + 8 * g;
        float d = 0.0f;                              // zero-pad K = 27..31
        if (k < KTAP) {
            const float px = lx + (float)(k / 9 - 1)       * DILF;
            const float py = ly + (float)((k / 3) % 3 - 1) * DILF;
            const float pz = lz + (float)(k % 3 - 1)       * DILF;
            d = MAXDF;
#pragma unroll
            for (int m = 0; m < NOBJ; ++m) {
                float vx = px - tx[m], vy = py - ty[m], vz = pz - tz[m];
                // inverse rotation: v + 2*u x (u x v + w*v), u = -q.xyz
                float cx = uy[m] * vz - uz[m] * vy;
                float cy = uz[m] * vx - ux[m] * vz;
                float cz = ux[m] * vy - uy[m] * vx;
                float t1x = cx + qw[m] * vx;
                float t1y = cy + qw[m] * vy;
                float t1z = cz + qw[m] * vz;
                float rx = (vx + 2.0f * (uy[m] * t1z - uz[m] * t1y)) * invs[m];
                float ry = (vy + 2.0f * (uz[m] * t1x - ux[m] * t1z)) * invs[m];
                float rz = (vz + 2.0f * (ux[m] * t1y - uy[m] * t1x)) * invs[m];
                bool inb = (rx >= 0.0f) & (rx <= hib[m]) &
                           (ry >= 0.0f) & (ry <= hib[m]) &
                           (rz >= 0.0f) & (rz <= hib[m]);
                float gx = fminf(fmaxf(rx * invc[m] - 0.5f, 0.0f), 63.0f);
                float gy = fminf(fmaxf(ry * invc[m] - 0.5f, 0.0f), 63.0f);
                float gz = fminf(fmaxf(rz * invc[m] - 0.5f, 0.0f), 63.0f);
                int ix = (int)fminf(floorf(gx), 62.0f);
                int iy = (int)fminf(floorf(gy), 62.0f);
                int iz = (int)fminf(floorf(gz), 62.0f);
                float fx = gx - (float)ix, fy = gy - (float)iy, fz = gz - (float)iz;
                const float* p0 = sdf + gb[m] + ((ix * GRID_N + iy) * GRID_N + iz);
                float v000 = p0[0];
                float v001 = p0[1];
                float v010 = p0[GRID_N];
                float v011 = p0[GRID_N + 1];
                float v100 = p0[GRID_N * GRID_N];
                float v101 = p0[GRID_N * GRID_N + 1];
                float v110 = p0[GRID_N * GRID_N + GRID_N];
                float v111 = p0[GRID_N * GRID_N + GRID_N + 1];
                float c00 = v000 + fz * (v001 - v000);
                float c01 = v010 + fz * (v011 - v010);
                float c10 = v100 + fz * (v101 - v100);
                float c11 = v110 + fz * (v111 - v110);
                float c0  = c00 + fy * (c01 - c00);
                float c1  = c10 + fy * (c11 - c10);
                float val = c0 + fx * (c1 - c0);
                val = inb ? val * scl[m] : MAXDF;
                d = fminf(d, val);
            }
        }
        dstage[myOff + i] = (_Float16)d;
    }

    // contiguous 32B reload -> ds_load_b128 pair, no dynamic VGPR indexing
    v16h afrag = *(const v16h*)(&dstage[myOff]);

    // ---- B fragments per 16-bit B (32x16) layout: k = i + (i>=8 ? 8 : 0) + 8*g, col = lane&15.
    //      Column permutation: tile0 -> even out channels (2*col),
    //                          tile1 -> odd  out channels (2*col+1)
    //      so each lane's two results are memory-adjacent (b64 stores).
    v16h bf0, bf1;
#pragma unroll
    for (int i = 0; i < 16; ++i) {
        const int kb = i + ((i >= 8) ? 8 : 0) + 8 * g;
        float w0 = (kb < KTAP) ? weight[(2 * col)     * KTAP + kb] : 0.0f;
        float w1 = (kb < KTAP) ? weight[(2 * col + 1) * KTAP + kb] : 0.0f;
        bf0[i] = (_Float16)w0;
        bf1[i] = (_Float16)w1;
    }

    // accumulators pre-loaded with bias (same channel for all 8 rows of a lane)
    v8f acc0, acc1;
    const float bs0 = bias[2 * col], bs1 = bias[2 * col + 1];
#pragma unroll
    for (int c = 0; c < 8; ++c) { acc0[c] = bs0; acc1[c] = bs1; }

    acc0 = __builtin_amdgcn_wmma_f32_16x16x32_f16(false, afrag, false, bf0,
                                                  (short)0, acc0, false, false);
    acc1 = __builtin_amdgcn_wmma_f32_16x16x32_f16(false, afrag, false, bf1,
                                                  (short)0, acc1, false, false);

    // ---- D (16x16 f32) layout: VGPR c -> row c + 8*g, col = lane&15.
    //      Row base hoisted once; 8x global_store_b64 at immediate offsets c*128B.
    float* orow = out + ((long)bb * npts + base + 8 * g) * NOUT + 2 * col;
    if (base + 16 <= npts) {                    // full tile: uniform fast path
#pragma unroll
        for (int c = 0; c < 8; ++c) {
            float2 v = make_float2(acc0[c], acc1[c]);
            *(float2*)(orow + c * NOUT) = v;
        }
    } else {                                    // ragged tail tile
#pragma unroll
        for (int c = 0; c < 8; ++c) {
            if (base + c + 8 * g < npts) {
                float2 v = make_float2(acc0[c], acc1[c]);
                *(float2*)(orow + c * NOUT) = v;
            }
        }
    }
}

extern "C" void kernel_launch(void* const* d_in, const int* in_sizes, int n_in,
                              void* d_out, int out_size, void* d_ws, size_t ws_size,
                              hipStream_t stream) {
    const float* locs   = (const float*)d_in[0];
    const int*   idxs   = (const int*)  d_in[1];
    const float* poses  = (const float*)d_in[2];
    const float* scales = (const float*)d_in[3];
    const float* sdf    = (const float*)d_in[4];
    const float* cells  = (const float*)d_in[5];
    const float* weight = (const float*)d_in[6];
    const float* bias   = (const float*)d_in[7];
    float* out = (float*)d_out;

    const int B = in_sizes[1] / NOBJ;        // idxs is (B, M)
    const int N = in_sizes[0] / (B * 3);     // locs is (B, N, 3)
    const int ntiles = (N + 15) / 16;        // one wave per 16-point tile
    dim3 grd((ntiles + WAVES_PB - 1) / WAVES_PB, B);
    ConvSDF_62191126446547_kernel<<<grd, WAVES_PB * 32, 0, stream>>>(
        locs, idxs, poses, scales, sdf, cells, weight, bias, out, N, ntiles);
}
